// THTN1_16552803959364
// MI455X (gfx1250) — matrix operand order repaired and checked by reading, not compiled, not run on hardware
//
#include <hip/hip_runtime.h>
#include <math.h>

// ---------------- problem constants ----------------
#define NNODES 50000
#define NEDGES 25000
#define NINC   800000
#define DIM    256           // V_DIM == E_DIM == Q_DIM
#define INDIM  512
#define NCLS   40
#define MLPD   128
#define LN_EPS 1e-5f

typedef __attribute__((ext_vector_type(16))) __bf16 v16bf;
typedef __attribute__((ext_vector_type(8)))  float  v8f;
typedef __attribute__((ext_vector_type(2)))  float  f32x2;
typedef __attribute__((ext_vector_type(2)))  __bf16 bf16x2;

// Packed f32 -> bf16x2 (RNE via fptrunc); should lower to native packed cvt.
__device__ __forceinline__ unsigned int pack_bf16x2(float lo, float hi) {
    f32x2 v; v.x = lo; v.y = hi;
    bf16x2 b = __builtin_convertvector(v, bf16x2);
    union { bf16x2 b; unsigned int u; } cv; cv.b = b;
    return cv.u;
}

__device__ __forceinline__ void atomicMaxF32(float* addr, float val) {
    int* ai = (int*)addr;
    int old = __float_as_int(*addr);
    while (__int_as_float(old) < val) {
        int assumed = old;
        old = atomicCAS(ai, assumed, __float_as_int(val));
        if (old == assumed) break;
    }
}

// ------------- bf16 WMMA GEMM: C[M,N] = A[M,K] @ B[K,N] + bias (opt ReLU) ----
// 128 threads = 4 waves. Block tile 64x64, wave tile 32x32 (4 WMMAs from
// 2 A-frags x 2 B-frags). f32->bf16 conversion fused into LDS staging.
// Double-buffered LDS; next tile's global loads overlap current WMMAs.
// Assumes A,B rows are 16B-aligned for full tiles (K,N multiples of 4).
#define BM 64
#define BN 64
#define BK 32

__global__ void __launch_bounds__(128)
gemm_bf16_wmma(const float* __restrict__ A, const float* __restrict__ B,
               const float* __restrict__ bias, float* __restrict__ C,
               int M, int N, int K, int do_relu)
{
    __shared__ unsigned short As[2][BM][BK + 2];   // [buf][m][k] bf16
    __shared__ unsigned short Bs[2][BN][BK + 2];   // [buf][n][k] (B transposed)

    const int tid  = threadIdx.x;
    const int lane = tid & 31;
    const int wave = tid >> 5;          // 0..3
    const int wm   = wave & 1;          // M sub-tile (32 rows)
    const int wn   = wave >> 1;         // N sub-tile (32 cols)
    const int m0   = blockIdx.x * BM;
    const int n0   = blockIdx.y * BN;
    const int idx  = lane & 15;
    const int h    = lane >> 4;

    // ---- staging ownership (loop-invariant guards, hoisted) ----
    const int a_r   = tid >> 1;                 // 0..63
    const int a_c   = (tid & 1) * 16;           // 0 | 16
    const int a_row = m0 + a_r;
    const bool aOK  = (a_row < M);
    const float* aBase = A + (size_t)a_row * K + a_c;

    const int b_k = (tid & 15) * 2;             // 0..30 (even)
    const int b_n = (tid >> 4) * 8;             // 0..56
    const bool bFull = (n0 + b_n + 8 <= N);
    const float* bBase = B + (size_t)b_k * N + n0 + b_n;

    float ar[16];                               // A stage: row a_row, 16 k
    float br0[8], br1[8];                       // B stage: rows b_k, b_k+1, 8 n

    auto loadA = [&](int kk) {
        if (aOK) {
#pragma unroll
            for (int j = 0; j < 4; ++j) {
                float4 v = *(const float4*)(aBase + kk + 4 * j);
                ar[4*j+0] = v.x; ar[4*j+1] = v.y; ar[4*j+2] = v.z; ar[4*j+3] = v.w;
            }
        } else {
#pragma unroll
            for (int j = 0; j < 16; ++j) ar[j] = 0.f;
        }
    };
    auto loadB = [&](int kk) {
        const float* p0 = bBase + (size_t)kk * N;
        const float* p1 = p0 + N;
        if (bFull) {
            float4 v0 = *(const float4*)(p0);
            float4 v1 = *(const float4*)(p0 + 4);
            float4 w0 = *(const float4*)(p1);
            float4 w1 = *(const float4*)(p1 + 4);
            br0[0]=v0.x; br0[1]=v0.y; br0[2]=v0.z; br0[3]=v0.w;
            br0[4]=v1.x; br0[5]=v1.y; br0[6]=v1.z; br0[7]=v1.w;
            br1[0]=w0.x; br1[1]=w0.y; br1[2]=w0.z; br1[3]=w0.w;
            br1[4]=w1.x; br1[5]=w1.y; br1[6]=w1.z; br1[7]=w1.w;
        } else {
#pragma unroll
            for (int j = 0; j < 8; ++j) {
                int n = n0 + b_n + j;
                br0[j] = (n < N) ? p0[j] : 0.f;
                br1[j] = (n < N) ? p1[j] : 0.f;
            }
        }
    };
    auto storeTiles = [&](int buf) {
#pragma unroll
        for (int j = 0; j < 8; ++j)
            *(unsigned int*)&As[buf][a_r][a_c + 2 * j] = pack_bf16x2(ar[2*j], ar[2*j+1]);
#pragma unroll
        for (int j = 0; j < 8; ++j)
            *(unsigned int*)&Bs[buf][b_n + j][b_k] = pack_bf16x2(br0[j], br1[j]);
    };

    v8f acc00 = {}, acc01 = {}, acc10 = {}, acc11 = {};

    loadA(0); loadB(0);
    int buf = 0;
    for (int kk = 0; kk < K; kk += BK) {
        storeTiles(buf);
        __syncthreads();
        if (kk + BK < K) { loadA(kk + BK); loadB(kk + BK); }  // overlap w/ WMMAs

        union U { v16bf v; unsigned int u[8]; } a0, a1, b0, b1;
#pragma unroll
        for (int j = 0; j < 8; ++j) {
            int kA = ((j < 4) ? (2 * j) : (2 * j + 8)) + 8 * h;  // A 16x32 layout
            int kB = ((j < 4) ? (2 * j) : (2 * j + 8)) + 8 * h;  // B 32x16 layout (same k order as A)
            a0.u[j] = *(const unsigned int*)&As[buf][wm * 32 + idx][kA];
            a1.u[j] = *(const unsigned int*)&As[buf][wm * 32 + 16 + idx][kA];
            b0.u[j] = *(const unsigned int*)&Bs[buf][wn * 32 + idx][kB];
            b1.u[j] = *(const unsigned int*)&Bs[buf][wn * 32 + 16 + idx][kB];
        }
        acc00 = __builtin_amdgcn_wmma_f32_16x16x32_bf16(false, a0.v, false, b0.v,
                                                        (short)0, acc00, false, false);
        acc01 = __builtin_amdgcn_wmma_f32_16x16x32_bf16(false, a0.v, false, b1.v,
                                                        (short)0, acc01, false, false);
        acc10 = __builtin_amdgcn_wmma_f32_16x16x32_bf16(false, a1.v, false, b0.v,
                                                        (short)0, acc10, false, false);
        acc11 = __builtin_amdgcn_wmma_f32_16x16x32_bf16(false, a1.v, false, b1.v,
                                                        (short)0, acc11, false, false);
        buf ^= 1;
    }

    // ---- epilogue: D layout row = i + 8*h, col = idx ----
    const int cA = n0 + wn * 32 + idx;
    const int cB = cA + 16;
    const float biasA = (cA < N) ? bias[cA] : 0.f;
    const float biasB = (cB < N) ? bias[cB] : 0.f;
#pragma unroll
    for (int i = 0; i < 8; ++i) {
        int r0 = m0 + wm * 32 + 8 * h + i;
        int r1 = r0 + 16;
        if (r0 < M) {
            if (cA < N) {
                float v = acc00[i] + biasA;
                if (do_relu) v = fmaxf(v, 0.f);
                C[(size_t)r0 * N + cA] = v;
            }
            if (cB < N) {
                float v = acc01[i] + biasB;
                if (do_relu) v = fmaxf(v, 0.f);
                C[(size_t)r0 * N + cB] = v;
            }
        }
        if (r1 < M) {
            if (cA < N) {
                float v = acc10[i] + biasA;
                if (do_relu) v = fmaxf(v, 0.f);
                C[(size_t)r1 * N + cA] = v;
            }
            if (cB < N) {
                float v = acc11[i] + biasB;
                if (do_relu) v = fmaxf(v, 0.f);
                C[(size_t)r1 * N + cB] = v;
            }
        }
    }
}

// ---------------- fill ----------------
__global__ void fill_kernel(float* __restrict__ p, float v, int n) {
    int i = blockIdx.x * blockDim.x + threadIdx.x;
    if (i < n) p[i] = v;
}

// ---- attention: one wave per incidence; 256-d dot, leaky_relu, seg-max ----
__global__ void __launch_bounds__(256)
attn_score_max(const float* __restrict__ Kmat, const int* __restrict__ idxK,
               const float* __restrict__ Qmat, const int* __restrict__ idxQ,
               const int* __restrict__ seg, float* __restrict__ score,
               float* __restrict__ segmax, int n_inc)
{
    int w = (blockIdx.x * blockDim.x + threadIdx.x) >> 5;
    int lane = threadIdx.x & 31;
    if (w >= n_inc) return;
    const float* kp = Kmat + (size_t)idxK[w] * DIM;
    const float* qp = Qmat + (size_t)idxQ[w] * DIM;
    float s = 0.f;
#pragma unroll
    for (int j = 0; j < 8; ++j) {
        int d = lane + j * 32;
        s += kp[d] * qp[d];
    }
#pragma unroll
    for (int off = 16; off > 0; off >>= 1) s += __shfl_down(s, off, 32);
    if (lane == 0) {
        s = (s > 0.f ? s : 0.01f * s) * 0.0625f;   // leaky_relu * 1/sqrt(256)
        score[w] = s;
        atomicMaxF32(&segmax[seg[w]], s);
    }
}

__global__ void attn_exp_sum(const float* __restrict__ score, const int* __restrict__ seg,
                             const float* __restrict__ segmax, float* __restrict__ ex,
                             float* __restrict__ segsum, int n)
{
    int i = blockIdx.x * blockDim.x + threadIdx.x;
    if (i >= n) return;
    int s = seg[i];
    float e = expf(score[i] - segmax[s]);
    ex[i] = e;
    atomicAdd(&segsum[s], e);
}

__global__ void __launch_bounds__(256)
attn_scatter(const float* __restrict__ ex, const float* __restrict__ segsum,
             const int* __restrict__ seg, const int* __restrict__ idxV,
             const float* __restrict__ V, float* __restrict__ H, int n_inc)
{
    int w = (blockIdx.x * blockDim.x + threadIdx.x) >> 5;
    int lane = threadIdx.x & 31;
    if (w >= n_inc) return;
    int s = seg[w];
    float wgt = ex[w] / segsum[s];
    const float* vp = V + (size_t)idxV[w] * DIM;
    float* hp = H + (size_t)s * DIM;
#pragma unroll
    for (int j = 0; j < 8; ++j) {
        int d = lane + j * 32;
        atomicAdd(&hp[d], wgt * vp[d]);
    }
}

// ---- LayerNorm over 256 with optional residual: out = LN(a [+ b]) * g + beta ----
__global__ void __launch_bounds__(256)
layernorm_kernel(const float* __restrict__ a, const float* __restrict__ b,
                 const float* __restrict__ g, const float* __restrict__ beta,
                 float* __restrict__ out, int rows)
{
    int w = (blockIdx.x * blockDim.x + threadIdx.x) >> 5;
    int lane = threadIdx.x & 31;
    if (w >= rows) return;
    const float* ap = a + (size_t)w * DIM;
    const float* bp = b ? b + (size_t)w * DIM : nullptr;
    float x[8];
    float sum = 0.f;
#pragma unroll
    for (int j = 0; j < 8; ++j) {
        int d = lane + j * 32;
        float v = ap[d];
        if (bp) v += bp[d];
        x[j] = v;
        sum += v;
    }
#pragma unroll
    for (int off = 16; off > 0; off >>= 1) sum += __shfl_xor(sum, off, 32);
    float mu = sum * (1.0f / DIM);
    float vs = 0.f;
#pragma unroll
    for (int j = 0; j < 8; ++j) { float t = x[j] - mu; vs += t * t; }
#pragma unroll
    for (int off = 16; off > 0; off >>= 1) vs += __shfl_xor(vs, off, 32);
    float inv = rsqrtf(vs * (1.0f / DIM) + LN_EPS);
    float* op = out + (size_t)w * DIM;
#pragma unroll
    for (int j = 0; j < 8; ++j) {
        int d = lane + j * 32;
        op[d] = (x[j] - mu) * inv * g[d] + beta[d];
    }
}

// ---------------- host orchestration ----------------
extern "C" void kernel_launch(void* const* d_in, const int* in_sizes, int n_in,
                              void* d_out, int out_size, void* d_ws, size_t ws_size,
                              hipStream_t stream) {
    (void)in_sizes; (void)n_in; (void)out_size; (void)ws_size;

    const float* vfeat = (const float*)d_in[0];
    const float* efeat = (const float*)d_in[1];
    const int*   src   = (const int*)d_in[2];
    const int*   dst   = (const int*)d_in[3];
    const float* W_v1 = (const float*)d_in[4];  const float* b_v1 = (const float*)d_in[5];
    const float* W_kv = (const float*)d_in[6];  const float* b_kv = (const float*)d_in[7];
    const float* W_vv = (const float*)d_in[8];  const float* b_vv = (const float*)d_in[9];
    const float* W_qe = (const float*)d_in[10]; const float* b_qe = (const float*)d_in[11];
    const float* W_ke = (const float*)d_in[12]; const float* b_ke = (const float*)d_in[13];
    const float* W_ve = (const float*)d_in[14]; const float* b_ve = (const float*)d_in[15];
    const float* W_qv = (const float*)d_in[16]; const float* b_qv = (const float*)d_in[17];
    const float* W_f1 = (const float*)d_in[18]; const float* b_f1 = (const float*)d_in[19];
    const float* W_f2 = (const float*)d_in[20]; const float* b_f2 = (const float*)d_in[21];
    const float* W_f3 = (const float*)d_in[22]; const float* b_f3 = (const float*)d_in[23];
    const float* W_f4 = (const float*)d_in[24]; const float* b_f4 = (const float*)d_in[25];
    const float* W_cls= (const float*)d_in[26]; const float* b_cls= (const float*)d_in[27];
    const float* W_mlp= (const float*)d_in[28]; const float* b_mlp= (const float*)d_in[29];
    const float* ln1_g = (const float*)d_in[30]; const float* ln1_b = (const float*)d_in[31];
    const float* ln2_g = (const float*)d_in[32]; const float* ln2_b = (const float*)d_in[33];

    float* out = (float*)d_out;

    // ---- workspace layout (floats); lifetimes allow aliasing. ~417 MB ----
    const size_t SZ_V = (size_t)NNODES * DIM;   // 12.8M
    const size_t SZ_E = (size_t)NEDGES * DIM;   // 6.4M
    float* W = (float*)d_ws;
    float* feat_v = W;                    // later reused as feat_v2
    float* k_v    = W + SZ_V;             // later h_v
    float* v_v    = W + 2 * SZ_V;         // later y
    float* q_v    = W + 3 * SZ_V;         // later mid2
    float* ffn2   = W + 4 * SZ_V;
    float* q_e    = W + 5 * SZ_V;               // later k_e
    float* h_e    = W + 5 * SZ_V + SZ_E;        // later v_e
    float* x_e    = W + 5 * SZ_V + 2 * SZ_E;
    float* mid_e  = W + 5 * SZ_V + 3 * SZ_E;
    float* ffn_e  = W + 5 * SZ_V + 4 * SZ_E;
    float* feat_e = W + 5 * SZ_V + 5 * SZ_E;
    float* score  = W + 5 * SZ_V + 6 * SZ_E;
    float* ex     = score + NINC;
    float* segmax = ex + NINC;
    float* segsum = segmax + NNODES;
    // aliases
    float* feat_v2 = feat_v;
    float* h_v  = k_v;
    float* y    = v_v;
    float* mid2 = q_v;
    float* k_e  = q_e;
    float* v_e  = h_e;

    auto gemm = [&](const float* A, const float* Bm, const float* bias, float* C,
                    int M, int N, int K, int relu) {
        dim3 g((M + BM - 1) / BM, (N + BN - 1) / BN);
        gemm_bf16_wmma<<<g, dim3(128), 0, stream>>>(A, Bm, bias, C, M, N, K, relu);
    };
    auto fill = [&](float* p, float v, size_t n) {
        fill_kernel<<<(int)((n + 255) / 256), 256, 0, stream>>>(p, v, (int)n);
    };
    const int incBlocks  = (NINC * 32 + 255) / 256;   // wave-per-incidence
    const int incThreads = (NINC + 255) / 256;

    // ---- node feature projection + attention-1 projections ----
    gemm(vfeat,  W_v1, b_v1, feat_v, NNODES, DIM, INDIM, 0);
    gemm(feat_v, W_kv, b_kv, k_v,    NNODES, DIM, DIM, 0);
    gemm(feat_v, W_vv, b_vv, v_v,    NNODES, DIM, DIM, 0);
    gemm(efeat,  W_qe, b_qe, q_e,    NEDGES, DIM, DIM, 0);

    // ---- attention 1: nodes -> hyperedges (segments = dst_edges) ----
    fill(segmax, -3.0e38f, NEDGES);
    fill(segsum, 0.f, NEDGES);
    fill(h_e, 0.f, SZ_E);
    attn_score_max<<<incBlocks, 256, 0, stream>>>(k_v, src, q_e, dst, dst, score, segmax, NINC);
    attn_exp_sum<<<incThreads, 256, 0, stream>>>(score, dst, segmax, ex, segsum, NINC);
    attn_scatter<<<incBlocks, 256, 0, stream>>>(ex, segsum, dst, src, v_v, h_e, NINC);

    // ---- edge-side add&norm + FFN ----
    const int lnBlocksE = (NEDGES * 32 + 255) / 256;
    const int lnBlocksV = (NNODES * 32 + 255) / 256;
    layernorm_kernel<<<lnBlocksE, 256, 0, stream>>>(h_e, efeat, ln1_g, ln1_b, x_e, NEDGES);
    gemm(x_e,   W_f1, b_f1, mid_e, NEDGES, DIM, DIM, 1);
    gemm(mid_e, W_f2, b_f2, ffn_e, NEDGES, DIM, DIM, 0);
    layernorm_kernel<<<lnBlocksE, 256, 0, stream>>>(ffn_e, x_e, ln1_g, ln1_b, feat_e, NEDGES);

    // ---- attention-2 projections ----
    gemm(feat_e, W_ke, b_ke, k_e, NEDGES, DIM, DIM, 0);
    gemm(feat_e, W_ve, b_ve, v_e, NEDGES, DIM, DIM, 0);
    gemm(feat_v, W_qv, b_qv, q_v, NNODES, DIM, DIM, 0);

    // ---- attention 2: hyperedges -> nodes (segments = src_nodes) ----
    fill(segmax, -3.0e38f, NNODES);
    fill(segsum, 0.f, NNODES);
    fill(h_v, 0.f, SZ_V);
    attn_score_max<<<incBlocks, 256, 0, stream>>>(k_e, dst, q_v, src, src, score, segmax, NINC);
    attn_exp_sum<<<incThreads, 256, 0, stream>>>(score, src, segmax, ex, segsum, NINC);
    attn_scatter<<<incBlocks, 256, 0, stream>>>(ex, segsum, src, dst, v_e, h_v, NINC);

    // ---- node-side add&norm + FFN ----
    layernorm_kernel<<<lnBlocksV, 256, 0, stream>>>(h_v, feat_v, ln2_g, ln2_b, y, NNODES);
    gemm(y,    W_f3, b_f3, mid2, NNODES, DIM, DIM, 1);
    gemm(mid2, W_f4, b_f4, ffn2, NNODES, DIM, DIM, 0);
    layernorm_kernel<<<lnBlocksV, 256, 0, stream>>>(ffn2, y, ln2_g, ln2_b, feat_v2, NNODES);

    // ---- heads: outputs concatenated (pred, feat_out) ----
    gemm(feat_v2, W_cls, b_cls, out,                         NNODES, NCLS, DIM, 0);
    gemm(feat_v2, W_mlp, b_mlp, out + (size_t)NNODES * NCLS, NNODES, MLPD, DIM, 0);
}
